// LiquidNN_76510547411481
// MI455X (gfx1250) — matrix operand order repaired and checked by reading, not compiled, not run on hardware
//
#include <hip/hip_runtime.h>
#include <hip/hip_bf16.h>

// ---------------------------------------------------------------------------
// Liquid NN recurrent cell on gfx1250 (MI455X), wave32 + WMMA bf16.
//
// 16 persistent blocks (one 16-row batch tile each; batch rows are
// independent recurrences). State x[16,1024] lives in f32 registers in WMMA
// C-fragment layout (wave w owns hidden cols [64w,64w+64)). GEMM inputs bf16,
// f32 accumulate. Weights pre-converted to bf16 scratch (3.1 MB, L2-resident
// against 512 re-reads). u_t staged one step ahead via CDNA5 async
// global->LDS copies (ASYNCcnt). Hardware v_tanh_f32 when available.
// ---------------------------------------------------------------------------

typedef __attribute__((ext_vector_type(16))) __bf16 v16bf;
typedef __attribute__((ext_vector_type(8)))  __bf16 v8bf;
typedef __attribute__((ext_vector_type(8)))  float  v8f;

union BF16x16 { v16bf v; v8bf h[2]; };

constexpr int IN_DIM = 256;
constexpr int HID    = 1024;
constexpr int OUT    = 256;
constexpr int BATCH  = 256;
constexpr int TSTEPS = 512;
constexpr int K1     = HID + IN_DIM;   // 1280
constexpr int LDST   = K1 + 8;         // 1288 halves: 644 dwords == 4 mod 64 banks
constexpr float DT_C = 0.1f;

static __device__ __forceinline__ v8f splat8(float v) {
  v8f r;
#pragma unroll
  for (int i = 0; i < 8; ++i) r[i] = v;
  return r;
}

#if __has_builtin(__builtin_amdgcn_tanhf)
static __device__ __forceinline__ float fast_tanh(float x) {
  return __builtin_amdgcn_tanhf(x);            // v_tanh_f32 (CDNA5 trans op)
}
#else
static __device__ __forceinline__ float fast_tanh(float x) {
  // branch-free: tanh(x) = sign(x) * (1 - e) / (1 + e), e = 2^(-2*log2(e)*|x|)
  float ax = __builtin_fabsf(x);
  float e  = __builtin_amdgcn_exp2f(ax * -2.885390081777927f);
  float r  = (1.0f - e) * __builtin_amdgcn_rcpf(1.0f + e);
  return __builtin_copysignf(r, x);
}
#endif

// CDNA5 async copy: 16 bytes global -> LDS, tracked by ASYNCcnt (no VGPR data).
static __device__ __forceinline__ void async_g2l_b128(unsigned lds_addr,
                                                      const void* gaddr) {
  asm volatile("global_load_async_to_lds_b128 %0, %1, off"
               :: "v"(lds_addr), "v"((unsigned long long)(uintptr_t)gaddr)
               : "memory");
}
static __device__ __forceinline__ void wait_asynccnt0() {
  asm volatile("s_wait_asynccnt 0x0" ::: "memory");
}
// Generic LDS pointer -> 32-bit LDS offset (flat LDS addresses carry the
// offset in their low 32 bits per CDNA5 aperture mapping).
static __device__ __forceinline__ unsigned lds_off(const void* p) {
  return (unsigned)(uintptr_t)p;
}

// --- f32 -> bf16 weight conversion (runs every launch; deterministic) -------
__global__ void liquid_cvt_bf16(const float* __restrict__ src,
                                __bf16* __restrict__ dst, int n) {
  int i = blockIdx.x * blockDim.x + threadIdx.x;
  if (i < n) dst[i] = (__bf16)src[i];
}

// --- persistent recurrent kernel --------------------------------------------
__global__ __launch_bounds__(512)
void liquid_persistent_kernel(const float* __restrict__ u_seq,   // [B][T][IN]
                              const float* __restrict__ theta,   // [HID]
                              const __bf16* __restrict__ Wfc,    // [HID][K1] bf16
                              const float* __restrict__ bfc,     // [HID]
                              const __bf16* __restrict__ Wro,    // [OUT][HID] bf16
                              const float* __restrict__ bro,     // [OUT]
                              float* __restrict__ out)           // [B][T][OUT]
{
  __shared__ __bf16 inp[16 * LDST];          // activation tile, 41216 B
  __shared__ float  ustage[16 * IN_DIM];     // async u_t staging, 16384 B

  const int tid  = threadIdx.x;
  const int wave = tid >> 5;          // 0..15
  const int lane = tid & 31;
  const int hl   = lane >> 4;         // lane-half select (WMMA layout)
  const int lm   = lane & 15;         // M for A, N for B/C
  const int m0   = blockIdx.x * 16;   // batch-row base

  // zero the x-part of the activation tile (x0 == 0)
  {
    v8bf z;
#pragma unroll
    for (int i = 0; i < 8; ++i) z[i] = (__bf16)0.0f;
    for (int i = tid; i < 16 * (HID / 8); i += blockDim.x) {
      int row = i / (HID / 8), c = (i % (HID / 8)) * 8;
      *(v8bf*)(&inp[row * LDST + c]) = z;
    }
  }

  // per-wave constants: wave owns hidden cols [wave*64, wave*64+64) (4 tiles)
  float theta_f[4], bfc_f[4];
#pragma unroll
  for (int j = 0; j < 4; ++j) {
    int n0 = (wave * 4 + j) * 16;
    theta_f[j] = theta[n0 + lm];
    bfc_f[j]   = bfc[n0 + lm];
  }
  const int   o0    = wave * 16;      // wave's GEMM2 output tile
  const float bro_f = bro[o0 + lm];

  // recurrent state in registers, C-fragment layout
  v8f xreg[4];
#pragma unroll
  for (int j = 0; j < 4; ++j) xreg[j] = splat8(0.0f);

  // each thread owns 8 floats of the u tile: row urow, cols [ucol, ucol+8)
  const int urow = tid >> 5;
  const int ucol = (tid & 31) * 8;
  const float*   u_base   = u_seq + ((size_t)(m0 + urow) * TSTEPS) * IN_DIM + ucol;
  float*         my_stage = &ustage[urow * IN_DIM + ucol];
  const unsigned my_lds   = lds_off(my_stage);

  // prologue: kick off async copy of u_0
  async_g2l_b128(my_lds,      u_base);
  async_g2l_b128(my_lds + 16, u_base + 4);

  __syncthreads();

  for (int t = 0; t < TSTEPS; ++t) {
    // ---- (1) consume staged u_t -> bf16 cols [1024,1280); prefetch u_{t+1} --
    {
      wait_asynccnt0();                       // this thread's region landed
      float4 a = *(const float4*)(my_stage);
      float4 b = *(const float4*)(my_stage + 4);
      v8bf ub;
      ub[0] = (__bf16)a.x; ub[1] = (__bf16)a.y; ub[2] = (__bf16)a.z; ub[3] = (__bf16)a.w;
      ub[4] = (__bf16)b.x; ub[5] = (__bf16)b.y; ub[6] = (__bf16)b.z; ub[7] = (__bf16)b.w;
      *(v8bf*)(&inp[urow * LDST + HID + ucol]) = ub;
      if (t + 1 < TSTEPS) {                   // async stage next step's u
        const float* up = u_base + (size_t)(t + 1) * IN_DIM;
        async_g2l_b128(my_lds,      up);
        async_g2l_b128(my_lds + 16, up + 4);
      }
    }
    __syncthreads();

    // ---- (2) GEMM1: acc = inp @ Wfc^T + bfc  (4 N-tiles per wave) ----
    v8f acc[4];
#pragma unroll
    for (int j = 0; j < 4; ++j) acc[j] = splat8(bfc_f[j]);

#pragma unroll 2
    for (int k0 = 0; k0 < K1; k0 += 32) {
      BF16x16 afrag;
      const __bf16* ap = &inp[lm * LDST + k0 + hl * 8];
      afrag.h[0] = *(const v8bf*)ap;          // K = off..off+7
      afrag.h[1] = *(const v8bf*)(ap + 16);   // K = off+16..off+23
#pragma unroll
      for (int j = 0; j < 4; ++j) {
        int n0 = (wave * 4 + j) * 16;
        const __bf16* bp = Wfc + (size_t)(n0 + lm) * K1 + k0 + (hl << 3);
        BF16x16 bfrag;
        bfrag.h[0] = *(const v8bf*)bp;        // K = base..base+7
        bfrag.h[1] = *(const v8bf*)(bp + 16); // K = base+16..base+23
        acc[j] = __builtin_amdgcn_wmma_f32_16x16x32_bf16(
            false, afrag.v, false, bfrag.v, (short)0, acc[j], false, false);
      }
    }

    // state update entirely in registers: x += DT*(tanh(acc) - theta*x)
#pragma unroll
    for (int j = 0; j < 4; ++j)
#pragma unroll
      for (int r = 0; r < 8; ++r) {
        float xv = xreg[j][r];
        xreg[j][r] = xv + DT_C * (fast_tanh(acc[j][r]) - theta_f[j] * xv);
      }
    __syncthreads();   // all A-tile reads done before overwriting x-part

    // ---- (3) publish bf16(x_new) into cols [0,1024) ----
#pragma unroll
    for (int j = 0; j < 4; ++j) {
      int n0 = (wave * 4 + j) * 16;
#pragma unroll
      for (int r = 0; r < 8; ++r) {
        int M = r + hl * 8;
        inp[M * LDST + n0 + lm] = (__bf16)xreg[j][r];
      }
    }
    __syncthreads();

    // ---- (4) GEMM2: out_t = x_new @ Wro^T + bro  (1 N-tile per wave) ----
    v8f acc2 = splat8(bro_f);
#pragma unroll 2
    for (int k0 = 0; k0 < HID; k0 += 32) {
      BF16x16 afrag;
      const __bf16* ap = &inp[lm * LDST + k0 + hl * 8];
      afrag.h[0] = *(const v8bf*)ap;
      afrag.h[1] = *(const v8bf*)(ap + 16);
      const __bf16* bp = Wro + (size_t)(o0 + lm) * HID + k0 + (hl << 3);
      BF16x16 bfrag;
      bfrag.h[0] = *(const v8bf*)bp;
      bfrag.h[1] = *(const v8bf*)(bp + 16);
      acc2 = __builtin_amdgcn_wmma_f32_16x16x32_bf16(
          false, afrag.v, false, bfrag.v, (short)0, acc2, false, false);
    }
#pragma unroll
    for (int r = 0; r < 8; ++r) {
      int M = r + hl * 8;
      out[((size_t)(m0 + M) * TSTEPS + t) * OUT + o0 + lm] = acc2[r];
    }
    // (4) reads only cols [0,1024); next (1) writes only cols [1024,1280)
    // and per-thread-owned ustage: disjoint, ordered by the loop barriers.
  }
}

// ---------------------------------------------------------------------------
extern "C" void kernel_launch(void* const* d_in, const int* in_sizes, int n_in,
                              void* d_out, int out_size, void* d_ws, size_t ws_size,
                              hipStream_t stream) {
  const float* u_seq = (const float*)d_in[0];   // [256][512][256]
  const float* theta = (const float*)d_in[1];   // [1024]
  const float* W_fc  = (const float*)d_in[2];   // [1024][1280]
  const float* b_fc  = (const float*)d_in[3];   // [1024]
  const float* W_ro  = (const float*)d_in[4];   // [256][1024]
  const float* b_ro  = (const float*)d_in[5];   // [256]
  float* out = (float*)d_out;                   // [256][512][256]

  const int nfc = HID * K1;      // 1,310,720
  const int nro = OUT * HID;     //   262,144
  __bf16* wfc_b = (__bf16*)d_ws;
  __bf16* wro_b = (__bf16*)((char*)d_ws + (size_t)nfc * sizeof(__bf16));

  liquid_cvt_bf16<<<(nfc + 255) / 256, 256, 0, stream>>>(W_fc, wfc_b, nfc);
  liquid_cvt_bf16<<<(nro + 255) / 256, 256, 0, stream>>>(W_ro, wro_b, nro);

  liquid_persistent_kernel<<<BATCH / 16, 512, 0, stream>>>(
      u_seq, theta, wfc_b, b_fc, wro_b, b_ro, out);
}
